// SkeletonResidual_43619687858800
// MI455X (gfx1250) — matrix operand order repaired and checked by reading, not compiled, not run on hardware
//
#include <hip/hip_runtime.h>

// ---------- types ----------
typedef __attribute__((ext_vector_type(16))) _Float16 v16h;
typedef __attribute__((ext_vector_type(8)))  _Float16 v8h;
typedef __attribute__((ext_vector_type(8)))  float    v8f;

union V16u { v16h v; v8h h[2]; };

// ---------- problem constants ----------
#define CIN    200
#define CINP   224      // Cin padded to 7*32 for K-chunks of 32 (t-major row width)
#define LROW   232      // LDS row stride in halves (pad vs 224 to spread banks)

// Skeleton adjacency (NEIGHBOUR_LIST) as bitmasks; input joint = ci/8, output joint = co/cpj.
__constant__ unsigned ADJ[25] = {
  0x1000337u, 0x0000011Fu, 0x0000000Fu, 0x0000000Eu,
  0x1000373u, 0x000001F1u, 0x000000F0u, 0x000000E0u,
  0x1000733u, 0x00000F11u, 0x00085F00u, 0x0018FE00u,
  0x0018FC00u, 0x00087800u, 0x0019FC00u, 0x000BD800u,
  0x0007C000u, 0x00078000u, 0x00070000u, 0x0038FC00u,
  0x00785800u, 0x00F80000u, 0x00F00000u, 0x00E00000u,
  0x1000111u };

// POOL_W: 14 pooled entries, each averaging 1 or 2 blocks of 16 channels
__constant__ int POOL_J0[14] = {0,2,4,6,8,10,12,14,15,17,19,20,22,24};
__constant__ int POOL_J1[14] = {1,3,5,7,9,11,13,-1,16,18,-1,21,23,-1};

// ---------- x: [B][200][T] f32  ->  xT: [B][T][224] f16 (zero-padded channels) ----------
__global__ __launch_bounds__(256) void transpose_cast(const float* __restrict__ x,
                                                      _Float16* __restrict__ xT) {
  __shared__ float tile[32][33];
  const int t0 = blockIdx.x * 32, c0 = blockIdx.y * 32, b = blockIdx.z;
  const int lj = threadIdx.x & 31, li = threadIdx.x >> 5;   // 8 rows per pass
  for (int p = 0; p < 4; ++p) {
    int c = c0 + li + p * 8;
    float v = 0.f;
    if (c < CIN) v = x[((size_t)b * CIN + c) * 8192 + t0 + lj];
    tile[li + p * 8][lj] = v;                                // tile[c_local][t_local]
  }
  __syncthreads();
  for (int p = 0; p < 4; ++p) {
    int tl = li + p * 8;
    xT[((size_t)b * 8192 + t0 + tl) * CINP + c0 + lj] = (_Float16)tile[lj][tl];
  }
}

// ---------- pack masked weights into WMMA A-fragment layout ----------
// Apack[mfrag][tap*7 + c7][lane 0..31][e 0..15]   (f16)
// A 16x32 f16 layout: M = lane%16 ; K = 8*(lane/16) + (e<8 ? e : e+8)
__global__ __launch_bounds__(256) void pack_weights(const float* __restrict__ w,
                                                    _Float16* __restrict__ apack,
                                                    int Cout, int cpj, int taps,
                                                    int mfrags) {
  int idx = blockIdx.x * 256 + threadIdx.x;
  int kch = taps * 7;
  int total = mfrags * kch * 32;
  if (idx >= total) return;
  int lane   = idx & 31;
  int kchunk = (idx >> 5) % kch;
  int mfrag  = idx / (32 * kch);
  int tap = kchunk / 7, c7 = kchunk % 7;
  int M  = mfrag * 16 + (lane & 15);
  int kb = 8 * (lane >> 4);
  _Float16* dst = apack + ((size_t)(mfrag * kch + kchunk) * 32 + lane) * 16;
  for (int e = 0; e < 16; ++e) {
    int kk = kb + (e < 8 ? e : e + 8);
    int ci = c7 * 32 + kk;
    float v = 0.f;
    if (M < Cout && ci < CIN) {
      int jo = M / cpj, ji = ci >> 3;
      if ((ADJ[jo] >> ji) & 1u) v = w[((size_t)M * CIN + ci) * taps + tap];
    }
    dst[e] = (_Float16)v;
  }
}

// ---------- fused masked conv1d as WMMA GEMM (t-major f16 input) ----------
// inT : f16 [B][Tin][224]   out: f16 [B][Tout][224] (OUT_F16) or f32 [B][Cout][Tout]
// Workgroup: 256 thr = 8 waves; wave w -> M-fragment mtile*8+w; N tile = 64 t.
template<int TAPS, int STRIDE, bool PRELU, bool OUT_F16>
__global__ __launch_bounds__(256)
void conv_wmma(const _Float16* __restrict__ inT, const _Float16* __restrict__ apack,
               const float* __restrict__ bias, const float* __restrict__ aslope,
               void* __restrict__ outp, int Cout, int cpj, int Tin, int Tout, int pad) {
  constexpr int NT   = 64;
  constexpr int KCH  = TAPS * 7;
  constexpr int SPAN = STRIDE * (NT - 1) + TAPS;        // input rows needed (<=129)
  __shared__ __align__(16) _Float16 xt[SPAN * LROW];    // [t][ci] tile

  const int t0    = blockIdx.x * NT;
  const int mtile = blockIdx.y;
  const int b     = blockIdx.z;
  const int wave  = threadIdx.x >> 5;
  const int lane  = threadIdx.x & 31;

  // ---- async-stage the (linear) t-major tile into LDS ----
  const int tbase = t0 * STRIDE - pad;
  int lo = tbase < 0 ? 0 : tbase;
  int hi = tbase + SPAN; if (hi > Tin) hi = Tin;
  const int rowOff  = lo - tbase;                        // rows to fix at front
  const int nrows   = hi - lo;
  const int nchunks = nrows * 28;                        // 28 x 16B per 448B row
  const unsigned xtb = (unsigned)(unsigned long long)(const void*)&xt[0];
  const unsigned long long gbase =
      (unsigned long long)(const void*)(inT + ((size_t)b * Tin + lo) * CINP);
  for (int ch = threadIdx.x; ch < nchunks; ch += 256) {
    int r = ch / 28, j = ch - r * 28;
    unsigned ldsa = xtb + (unsigned)((rowOff + r) * (LROW * 2) + j * 16);
    unsigned long long ga = gbase + (unsigned long long)r * (CINP * 2) + j * 16;
    asm volatile("global_load_async_to_lds_b128 %0, %1, off"
                 :: "v"(ldsa), "v"(ga) : "memory");
  }
  asm volatile("s_wait_asynccnt 0x0" ::: "memory");
  __syncthreads();

  // ---- reflect-pad fixups (block-uniform, at most 1 row per edge) ----
  if (rowOff > 0 || (hi - tbase) < SPAN) {
    for (int c = threadIdx.x; c < CINP; c += 256) {
      for (int r = 0; r < rowOff; ++r) {                 // g = tbase+r < 0 -> reflect -g
        int src = (-(tbase + r)) - tbase;
        xt[r * LROW + c] = xt[src * LROW + c];
      }
      for (int r = hi - tbase; r < SPAN; ++r) {          // g >= Tin -> reflect 2Tin-2-g
        int src = (2 * Tin - 2 - (tbase + r)) - tbase;
        xt[r * LROW + c] = xt[src * LROW + c];
      }
    }
    __syncthreads();
  }

  // ---- WMMA accumulation ----
  const int mfrags = (Cout + 15) >> 4;
  const int mfrag  = mtile * 8 + wave;
  const int mload  = (mfrag < mfrags) ? mfrag : 0;

  // joint-block sparsity: skip K-chunks with no adjacency (wave-uniform branch)
  int jo0 = (mload * 16) / cpj;        if (jo0 > 24) jo0 = 24;
  int jo1 = (mload * 16 + 15) / cpj;   if (jo1 > 24) jo1 = 24;
  const unsigned adjU = ADJ[jo0] | ADJ[jo1];

  v8f acc[4] = {};
  const _Float16* abase = apack + ((size_t)mload * KCH * 32 + lane) * 16;
  const int laneN  = lane & 15;
  const int ciHalf = 8 * (lane >> 4);

  for (int kc = 0; kc < KCH; ++kc) {
    const int tap = kc / 7;
    const int c7  = kc - tap * 7;
    if (((adjU >> (c7 * 4)) & 0xFu) == 0u) continue;

    V16u a;
    const _Float16* ap = abase + (size_t)kc * 512;
    a.h[0] = *(const v8h*)(ap);
    a.h[1] = *(const v8h*)(ap + 8);

    const int ci0 = c7 * 32 + ciHalf;                    // B: K = 8*(lane/16) + (e<8 ? e : e+8)
    for (int ns = 0; ns < 4; ++ns) {
      const int r = (ns * 16 + laneN) * STRIDE + tap;
      const _Float16* bp = &xt[r * LROW + ci0];
      V16u bb;
      bb.h[0] = *(const v8h*)(bp);
      bb.h[1] = *(const v8h*)(bp + 16);
      acc[ns] = __builtin_amdgcn_wmma_f32_16x16x32_f16(
          false, a.v, false, bb.v, (short)0, acc[ns], false, false);
    }
  }

  // ---- epilogue ----
  if (OUT_F16) {
    // bounce through LDS so global f16 stores are coalesced t-major rows
    __syncthreads();                                     // done reading xt
    _Float16* hout = xt;                                 // 64 x 128 f16 (16KB) reuse
    const int chl = wave * 16 + 8 * (lane >> 4);
    const float sl = PRELU ? aslope[0] : 0.f;
    for (int ns = 0; ns < 4; ++ns) {
      const int tl = ns * 16 + laneN;
      for (int i = 0; i < 8; ++i) {
        const int M = mtile * 128 + chl + i;
        float v = 0.f;
        if (M < Cout) {
          v = acc[ns][i] + bias[M];
          if (PRELU) v = (v >= 0.f) ? v : sl * v;
        }
        hout[tl * 128 + chl + i] = (_Float16)v;
      }
    }
    __syncthreads();
    _Float16* oT = (_Float16*)outp;
    const int cb = mtile * 128;
    const int cw = (CINP - cb) < 128 ? (CINP - cb) : 128; // 128 or 96 cols
    for (int idx = threadIdx.x; idx < NT * cw; idx += 256) {
      int tl = idx / cw, cl = idx - tl * cw;
      oT[((size_t)b * Tout + t0 + tl) * CINP + cb + cl] = hout[tl * 128 + cl];
    }
  } else if (mfrag < mfrags) {
    const int mb = mfrag * 16 + 8 * (lane >> 4);         // D: M = vgpr + 8*(lane/16)
    float* og = (float*)outp;
    for (int ns = 0; ns < 4; ++ns) {
      const int t = t0 + ns * 16 + laneN;                // D: N = lane%16
      for (int i = 0; i < 8; ++i) {
        const int M = mb + i;
        if (M < Cout) og[((size_t)b * Cout + M) * Tout + t] = acc[ns][i] + bias[M];
      }
    }
  }
}

// ---------- GroupNorm statistics (deterministic block reduction) ----------
__global__ __launch_bounds__(256) void gn_stats(const float* __restrict__ h,
                                                float* __restrict__ stats) {
  const int b = blockIdx.x / 10, g = blockIdx.x % 10;
  const float* p = h + ((size_t)b * 400 + g * 40) * 4096;
  float s = 0.f, s2 = 0.f;
  for (int e = threadIdx.x; e < 40 * 4096; e += 256) {
    float v = p[e]; s += v; s2 += v * v;
  }
  __shared__ float rs[256], rq[256];
  rs[threadIdx.x] = s; rq[threadIdx.x] = s2;
  __syncthreads();
  for (int off = 128; off > 0; off >>= 1) {
    if (threadIdx.x < off) { rs[threadIdx.x] += rs[threadIdx.x + off];
                             rq[threadIdx.x] += rq[threadIdx.x + off]; }
    __syncthreads();
  }
  if (threadIdx.x == 0) {
    const float n   = 40.f * 4096.f;
    const float mu  = rs[0] / n;
    const float var = rq[0] / n - mu * mu;
    stats[(b * 10 + g) * 2 + 0] = mu;
    stats[(b * 10 + g) * 2 + 1] = rsqrtf(var + 1e-5f);
  }
}

// ---------- GN + shortcut + channel-pool + PReLU ----------
__global__ __launch_bounds__(256) void fuse_out(const float* __restrict__ h,
                                                const float* __restrict__ s,
                                                const float* __restrict__ stats,
                                                const float* __restrict__ gamma,
                                                const float* __restrict__ gbeta,
                                                const float* __restrict__ a1,
                                                float* __restrict__ out) {
  const int t = blockIdx.x * 256 + threadIdx.x;
  const int p = blockIdx.y, b = blockIdx.z;
  const int j0 = POOL_J0[p], j1 = POOL_J1[p];
  const float inv = (j1 >= 0) ? 0.5f : 1.0f;
  const float slope = a1[0];
  for (int c = 0; c < 16; ++c) {
    int ch0 = j0 * 16 + c, g0 = ch0 / 40;
    float mu0 = stats[(b * 10 + g0) * 2], r0 = stats[(b * 10 + g0) * 2 + 1];
    size_t i0 = ((size_t)b * 400 + ch0) * 4096 + t;
    float v = (h[i0] - mu0) * r0 * gamma[ch0] + gbeta[ch0] + s[i0];
    if (j1 >= 0) {
      int ch1 = j1 * 16 + c, g1 = ch1 / 40;
      float mu1 = stats[(b * 10 + g1) * 2], r1 = stats[(b * 10 + g1) * 2 + 1];
      size_t i1 = ((size_t)b * 400 + ch1) * 4096 + t;
      v += (h[i1] - mu1) * r1 * gamma[ch1] + gbeta[ch1] + s[i1];
    }
    v *= inv;
    out[((size_t)b * 224 + p * 16 + c) * 4096 + t] = (v >= 0.f) ? v : slope * v;
  }
}

// ---------- launch ----------
extern "C" void kernel_launch(void* const* d_in, const int* in_sizes, int n_in,
                              void* d_out, int out_size, void* d_ws, size_t ws_size,
                              hipStream_t stream) {
  const float* x    = (const float*)d_in[0];
  const float* w0   = (const float*)d_in[1];
  const float* b0   = (const float*)d_in[2];
  const float* a0   = (const float*)d_in[3];
  const float* w1   = (const float*)d_in[4];
  const float* b1   = (const float*)d_in[5];
  const float* gng  = (const float*)d_in[6];
  const float* gnb  = (const float*)d_in[7];
  const float* wsc  = (const float*)d_in[8];
  const float* bs   = (const float*)d_in[9];
  const float* a1   = (const float*)d_in[10];
  float* out = (float*)d_out;

  // workspace carve-up (~330 MB)
  char* w = (char*)d_ws;
  size_t o = 0;
  _Float16* xT  = (_Float16*)(w + o); o += 16ull * 8192 * 224 * 2;   // x, t-major f16
  _Float16* h0T = (_Float16*)(w + o); o += 16ull * 8192 * 224 * 2;   // conv0+prelu, t-major f16
  float*    h   = (float*)(w + o);    o += 16ull * 400 * 4096 * 4;   // conv1 out, f32
  float*    sb  = (float*)(w + o);    o += 16ull * 400 * 4096 * 4;   // shortcut, f32
  _Float16* ap0 = (_Float16*)(w + o); o += 13ull * 21 * 512 * 2;     // packed w0*MASK0
  _Float16* ap1 = (_Float16*)(w + o); o += 25ull * 21 * 512 * 2;     // packed w1*MASK1
  _Float16* apS = (_Float16*)(w + o); o += 25ull * 7 * 512 * 2;      // packed ws*MASK_SC
  float*    st  = (float*)(w + o);                                    // [16][10][2]

  transpose_cast<<<dim3(256, 7, 16), 256, 0, stream>>>(x, xT);

  pack_weights<<<dim3((13 * 21 * 32 + 255) / 256), 256, 0, stream>>>(w0,  ap0, 200,  8, 3, 13);
  pack_weights<<<dim3((25 * 21 * 32 + 255) / 256), 256, 0, stream>>>(w1,  ap1, 400, 16, 3, 25);
  pack_weights<<<dim3((25 *  7 * 32 + 255) / 256), 256, 0, stream>>>(wsc, apS, 400, 16, 1, 25);

  // conv0 (200->200, k3 s1 reflect-pad1) + PReLU -> h0T (f16, t-major)
  conv_wmma<3, 1, true, true><<<dim3(128, 2, 16), 256, 0, stream>>>(
      xT, ap0, b0, a0, (void*)h0T, 200, 8, 8192, 8192, 1);
  // conv1 (200->400, k3 s2 reflect-pad1) -> h (f32, channel-major)
  conv_wmma<3, 2, false, false><<<dim3(64, 4, 16), 256, 0, stream>>>(
      h0T, ap1, b1, nullptr, (void*)h, 400, 16, 8192, 4096, 1);
  // shortcut (200->400, k1 s2 no-pad) -> s (f32, channel-major)
  conv_wmma<1, 2, false, false><<<dim3(64, 4, 16), 256, 0, stream>>>(
      xT, apS, bs, nullptr, (void*)sb, 400, 16, 8192, 4096, 0);

  gn_stats<<<dim3(160), 256, 0, stream>>>(h, st);
  fuse_out<<<dim3(16, 14, 16), 256, 0, stream>>>(h, sb, st, gng, gnb, a1, out);
}
